// ChangeDetectionMamba_39633958208137
// MI455X (gfx1250) — matrix operand order repaired and checked by reading, not compiled, not run on hardware
//
#include <hip/hip_runtime.h>

// ---------------- problem constants (from reference) ----------------
#define DIM     512
#define E_      336
#define EPAD    384          // E padded to multiple of 64 for pipelined WMMA K-loop
#define DSTATE  16
#define DCONV   4
#define HEADS   6
#define HD      56           // E/HEADS
#define QKV3    1008         // 3*E
#define B_      8
#define L_      4096
#define NTOK    (B_*L_)      // 32768
#define KC      819          // int(L*0.2)
#define KCPAD   832          // KC rounded to multiple of 16

// N padding for register-blocked GEMM (64-column wave tiles)
#define NP_IN   384          // 336 -> 384
#define NP_QKV  1024         // 1008 -> 1024
#define NP_OUT  512          // already multiple of 64

typedef __bf16 bf16_t;
typedef __attribute__((ext_vector_type(16))) __bf16 v16bf;
typedef __attribute__((ext_vector_type(8)))  __bf16 v8bf;
typedef __attribute__((ext_vector_type(8)))  float  v8f;

// ---------------- elementwise: DyT + fp32->bf16 ----------------
__global__ void dyt_kernel(const float* __restrict__ x, const float* __restrict__ alpha,
                           const float* __restrict__ w, const float* __restrict__ b,
                           bf16_t* __restrict__ out, long n) {
  long i = (long)blockIdx.x * blockDim.x + threadIdx.x;
  if (i >= n) return;
  int c = (int)(i % DIM);
  float v = tanhf(alpha[0] * x[i]) * w[c] + b[c];
  out[i] = (bf16_t)v;
}

// ---------------- weight prep: W (K x N, f32) -> W^T (Npad x Kpad, bf16, zero pad) ----------------
__global__ void transpose_pad_kernel(const float* __restrict__ W, bf16_t* __restrict__ WT,
                                     int K, int N, int Kpad, int Npad) {
  long t = (long)blockIdx.x * blockDim.x + threadIdx.x;
  long total = (long)Npad * Kpad;
  if (t >= total) return;
  int n = (int)(t / Kpad);
  int k = (int)(t % Kpad);
  WT[t] = (k < K && n < N) ? (bf16_t)W[(size_t)k * N + n] : (bf16_t)0.0f;
}

// ---------------- core bf16 WMMA GEMM: C = A(MxK) * B(KxN) + bias ----------------
// A: row-major M x K (bf16); BT: row-major Npad x K (bf16, B transposed, zero-padded rows).
// K multiple of 64, M multiple of 64, Npad multiple of 64, N multiple of 16.
// Block = 128 threads (4 waves) -> 64M x 64N per block.
// Each wave: 16M x 64N via 4 accumulators; two-stage software pipeline
// (double-buffered fragments) so loads for chunk k+32 overlap WMMAs on chunk k.
__global__ void gemm_bf16_wmma(const bf16_t* __restrict__ A,
                               const bf16_t* __restrict__ BT,
                               const float*  __restrict__ bias,
                               float* __restrict__ C,
                               int M, int N, int K) {
  const int wave   = threadIdx.x >> 5;
  const int lane   = threadIdx.x & 31;     // wave32
  const int half   = lane >> 4;
  const int lq     = lane & 15;
  const int tileN0 = blockIdx.x * 64;
  const int tileM  = blockIdx.y * 64 + wave * 16;

  const bf16_t* arow = A + (size_t)(tileM + lq) * K;          // A row m = lq
  const bf16_t* brow[4];
#pragma unroll
  for (int j = 0; j < 4; ++j)
    brow[j] = BT + (size_t)(tileN0 + 16 * j + lq) * K;        // B col n = lq of tile j

  v8f acc[4];
#pragma unroll
  for (int j = 0; j < 4; ++j) acc[j] = (v8f){};

  // Fragment loader for one 32-wide K chunk.
  // A fragment (16x32): lanes 0-15 -> K [k,k+8) & [k+16,k+24);
  //                     lanes 16-31 -> K [k+8,k+16) & [k+24,k+32)
  // B fragment (32x16): same K mapping as A
  auto load_frags = [&](int k, v16bf& af, v16bf (&bfv)[4]) {
    v8bf a0 = *(const v8bf*)(arow + k + half * 8);
    v8bf a1 = *(const v8bf*)(arow + k + 16 + half * 8);
#pragma unroll
    for (int i = 0; i < 8; ++i) { af[i] = a0[i]; af[i + 8] = a1[i]; }
#pragma unroll
    for (int j = 0; j < 4; ++j) {
      v8bf b0 = *(const v8bf*)(brow[j] + k + half * 8);
      v8bf b1 = *(const v8bf*)(brow[j] + k + 16 + half * 8);
#pragma unroll
      for (int i = 0; i < 8; ++i) { bfv[j][i] = b0[i]; bfv[j][i + 8] = b1[i]; }
    }
  };

  v16bf af0, af1;
  v16bf bf0[4], bf1[4];
  load_frags(0, af0, bf0);
  for (int k = 0; k < K; k += 64) {
    // stage 1: issue loads for k+32 (always in-bounds: K multiple of 64)
    load_frags(k + 32, af1, bf1);
    if (k + 64 < K) __builtin_prefetch(arow + k + 64, 0, 1);   // uniform
#pragma unroll
    for (int j = 0; j < 4; ++j)
      acc[j] = __builtin_amdgcn_wmma_f32_16x16x32_bf16(
          false, af0, false, bf0[j], (short)0, acc[j], false, false);
    // stage 2: issue loads for k+64 while consuming buffer 1
    if (k + 64 < K) load_frags(k + 64, af0, bf0);              // uniform guard
#pragma unroll
    for (int j = 0; j < 4; ++j)
      acc[j] = __builtin_amdgcn_wmma_f32_16x16x32_bf16(
          false, af1, false, bf1[j], (short)0, acc[j], false, false);
  }

  // D layout: element r -> M = r + 8*half, N = lq. Guarded (wave-uniform) edge stores.
#pragma unroll
  for (int j = 0; j < 4; ++j) {
    int tn = tileN0 + 16 * j;
    if (tn < N) {
      float bs = bias ? bias[tn + lq] : 0.0f;
#pragma unroll
      for (int r = 0; r < 8; ++r) {
        int m = tileM + r + half * 8;
        C[(size_t)m * N + tn + lq] = acc[j][r] + bs;
      }
    }
  }
}

// ---------------- row L2-normalize (wave per row), write bf16 padded ----------------
__global__ void rownorm_kernel(const float* __restrict__ xp, bf16_t* __restrict__ out) {
  int row = blockIdx.x, lane = threadIdx.x;
  const float* r = xp + (size_t)row * E_;
  float s = 0.f;
  for (int c = lane; c < E_; c += 32) { float v = r[c]; s += v * v; }
#pragma unroll
  for (int off = 16; off > 0; off >>= 1) s += __shfl_xor(s, off, 32);
  float inv = 1.0f / fmaxf(sqrtf(s), 1e-12f);
  bf16_t* o = out + (size_t)row * EPAD;
  for (int c = lane; c < EPAD; c += 32)
    o[c] = (c < E_) ? (bf16_t)(r[c] * inv) : (bf16_t)0.0f;
}

// ---------------- top-1 sparse attention: per (token, head) argmax + v gather ----------------
// out channel layout matches reference transpose: E index = d*HEADS + h
__global__ void attn_kernel(const float* __restrict__ qkv, float* __restrict__ out) {
  long t = (long)blockIdx.x * blockDim.x + threadIdx.x;
  if (t >= (long)NTOK * HEADS) return;
  int n = (int)(t / HEADS), h = (int)(t % HEADS);
  const float* base = qkv + (size_t)n * QKV3;
  const float* q  = base + h * HD;
  const float* k0 = base + E_;
  int best = 0; float bestv = -1e30f;
  for (int g = 0; g < HEADS; ++g) {
    const float* kg = k0 + g * HD;
    float s = 0.f;
    for (int d = 0; d < HD; ++d) s += q[d] * kg[d];
    if (s > bestv) { bestv = s; best = g; }
  }
  const float* v = base + 2 * E_ + best * HD;
  float* o = out + (size_t)n * E_;
  for (int d = 0; d < HD; ++d) o[(size_t)d * HEADS + h] = v[d];
}

// ---------------- importance = row L2 norm of attn out ----------------
__global__ void imp_kernel(const float* __restrict__ a, float* __restrict__ imp) {
  int row = blockIdx.x, lane = threadIdx.x;
  const float* r = a + (size_t)row * E_;
  float s = 0.f;
  for (int c = lane; c < E_; c += 32) { float v = r[c]; s += v * v; }
#pragma unroll
  for (int off = 16; off > 0; off >>= 1) s += __shfl_xor(s, off, 32);
  if (lane == 0) imp[row] = sqrtf(s);
}

// ---------------- per-batch iterative top-KC selection (sorted descending) ----------------
__global__ void topk_kernel(const float* __restrict__ imp, int* __restrict__ idx) {
  __shared__ float vals[L_];
  __shared__ float rv[256];
  __shared__ int   ri[256];
  int b = blockIdx.x, tid = threadIdx.x;
  const float* row = imp + (size_t)b * L_;
  for (int j = tid; j < L_; j += 256) vals[j] = row[j];
  __syncthreads();
  for (int i = 0; i < KC; ++i) {
    float bv = -1e30f; int bi = 1 << 30;
    for (int j = tid; j < L_; j += 256) {
      float v = vals[j];
      if (v > bv || (v == bv && j < bi)) { bv = v; bi = j; }
    }
    rv[tid] = bv; ri[tid] = bi;
    __syncthreads();
    for (int s = 128; s > 0; s >>= 1) {
      if (tid < s) {
        if (rv[tid + s] > rv[tid] || (rv[tid + s] == rv[tid] && ri[tid + s] < ri[tid])) {
          rv[tid] = rv[tid + s]; ri[tid] = ri[tid + s];
        }
      }
      __syncthreads();
    }
    if (tid == 0) { idx[b * KC + i] = ri[0]; vals[ri[0]] = -1e30f; }
    __syncthreads();
  }
}

// ---------------- fused depthwise causal conv4 + SSM scan; ys -> bf16 padded ----------------
__global__ void scan_kernel(const float* __restrict__ attn_out, const int* __restrict__ idx,
                            const float* __restrict__ conv_w, const float* __restrict__ A,
                            const float* __restrict__ Bp, const float* __restrict__ Cp,
                            bf16_t* __restrict__ ysh) {
  __shared__ float sA[DSTATE * DSTATE];
  __shared__ float sB[DSTATE];
  int b = blockIdx.x;
  int e = threadIdx.x;                              // 0..EPAD-1 (384 = 12 waves)
  if (e < DSTATE * DSTATE) sA[e] = A[e];
  if (e < DSTATE) sB[e] = 1.0f / (1.0f + expf(-Bp[e]));
  __syncthreads();
  bf16_t* ybase = ysh + (size_t)b * KCPAD * EPAD;
  if (e >= E_) {                                    // zero pad columns
    for (int t = 0; t < KCPAD; ++t) ybase[(size_t)t * EPAD + e] = (bf16_t)0.0f;
    return;
  }
  float w0 = conv_w[e * DCONV + 0], w1 = conv_w[e * DCONV + 1];
  float w2 = conv_w[e * DCONV + 2], w3 = conv_w[e * DCONV + 3];
  float sc[DSTATE];
#pragma unroll
  for (int i = 0; i < DSTATE; ++i) sc[i] = 1.0f / (1.0f + expf(-Cp[e * DSTATE + i]));
  float h[DSTATE];
#pragma unroll
  for (int i = 0; i < DSTATE; ++i) h[i] = 0.0f;
  float x0 = 0.f, x1 = 0.f, x2 = 0.f, x3 = 0.f;
  const int* bidx = idx + b * KC;
  for (int t = 0; t < KC; ++t) {
    int tok = bidx[t];
    float cur = attn_out[((size_t)b * L_ + tok) * E_ + e];   // coalesced over e
    x0 = x1; x1 = x2; x2 = x3; x3 = cur;
    float xt = w0 * x0 + w1 * x1 + w2 * x2 + w3 * x3;        // causal conv4
    float hn[DSTATE]; float y = 0.f;
#pragma unroll
    for (int i = 0; i < DSTATE; ++i) {
      float acc = sB[i] * xt;
#pragma unroll
      for (int j = 0; j < DSTATE; ++j) acc += sA[i * DSTATE + j] * h[j];
      hn[i] = acc; y += acc * sc[i];
    }
#pragma unroll
    for (int i = 0; i < DSTATE; ++i) h[i] = hn[i];
    ybase[(size_t)t * EPAD + e] = (bf16_t)y;
  }
  for (int t = KC; t < KCPAD; ++t) ybase[(size_t)t * EPAD + e] = (bf16_t)0.0f;  // zero pad rows
}

// ---------------- residual copy + scatter ----------------
__global__ void copy_kernel(const float* __restrict__ x, float* __restrict__ out, long n) {
  long i = (long)blockIdx.x * blockDim.x + threadIdx.x;
  if (i < n) out[i] = x[i];
}

__global__ void scatter_kernel(const float* __restrict__ xproc, const int* __restrict__ idx,
                               const float* __restrict__ x, float* __restrict__ out) {
  long t = (long)blockIdx.x * blockDim.x + threadIdx.x;
  if (t >= (long)B_ * KC * DIM) return;
  int c = (int)(t % DIM);
  long r = t / DIM;
  int i = (int)(r % KC);
  int b = (int)(r / KC);
  int tok = idx[b * KC + i];
  size_t o = ((size_t)b * L_ + tok) * DIM + c;
  out[o] = xproc[((size_t)b * KCPAD + i) * DIM + c] + x[o];
}

// ---------------- launcher ----------------
static inline size_t alignup(size_t v) { return (v + 255) & ~(size_t)255; }

extern "C" void kernel_launch(void* const* d_in, const int* in_sizes, int n_in,
                              void* d_out, int out_size, void* d_ws, size_t ws_size,
                              hipStream_t stream) {
  (void)in_sizes; (void)n_in; (void)out_size; (void)ws_size;
  const float* x     = (const float*)d_in[0];
  const float* alpha = (const float*)d_in[1];
  const float* dyt_w = (const float*)d_in[2];
  const float* dyt_b = (const float*)d_in[3];
  const float* W_in  = (const float*)d_in[4];
  const float* b_in  = (const float*)d_in[5];
  const float* W_qkv = (const float*)d_in[6];
  const float* b_qkv = (const float*)d_in[7];
  const float* convw = (const float*)d_in[8];
  const float* A     = (const float*)d_in[9];
  const float* Bp    = (const float*)d_in[10];
  const float* Cp    = (const float*)d_in[11];
  const float* W_out = (const float*)d_in[12];
  const float* b_out = (const float*)d_in[13];
  float* out = (float*)d_out;

  // workspace carve-out (with aliasing: attn_out<-xp, ys<-xpn, xproc<-qkv)
  char* ws = (char*)d_ws;
  size_t off = 0;
  bf16_t* xn_h   = (bf16_t*)(ws + off); off = alignup(off + (size_t)NTOK * DIM * 2);
  bf16_t* WinT   = (bf16_t*)(ws + off); off = alignup(off + (size_t)NP_IN * DIM * 2);
  bf16_t* WqkvT  = (bf16_t*)(ws + off); off = alignup(off + (size_t)NP_QKV * EPAD * 2);
  bf16_t* WoutT  = (bf16_t*)(ws + off); off = alignup(off + (size_t)NP_OUT * EPAD * 2);
  float*  xp     = (float*)(ws + off);  off = alignup(off + (size_t)NTOK * E_ * 4);
  bf16_t* xpn_h  = (bf16_t*)(ws + off); off = alignup(off + (size_t)NTOK * EPAD * 2);
  float*  qkv    = (float*)(ws + off);  off = alignup(off + (size_t)NTOK * QKV3 * 4);
  float*  imp    = (float*)(ws + off);  off = alignup(off + (size_t)NTOK * 4);
  int*    idx    = (int*)(ws + off);    off = alignup(off + (size_t)B_ * KC * 4);
  float*  attn_o = xp;                  // alias (xp dead after rownorm)
  bf16_t* ys_h   = xpn_h;               // alias (xpn dead after GEMM2)
  float*  xproc  = qkv;                 // alias (qkv dead after attn)

  // 1) DyT + bf16
  {
    long n = (long)NTOK * DIM;
    dyt_kernel<<<dim3((unsigned)((n + 255) / 256)), 256, 0, stream>>>(x, alpha, dyt_w, dyt_b, xn_h, n);
  }
  // 2) weight prep (transpose + pad + bf16)
  {
    long n1 = (long)NP_IN * DIM;
    transpose_pad_kernel<<<dim3((unsigned)((n1 + 255) / 256)), 256, 0, stream>>>(W_in, WinT, DIM, E_, DIM, NP_IN);
    long n2 = (long)NP_QKV * EPAD;
    transpose_pad_kernel<<<dim3((unsigned)((n2 + 255) / 256)), 256, 0, stream>>>(W_qkv, WqkvT, E_, QKV3, EPAD, NP_QKV);
    long n3 = (long)NP_OUT * EPAD;
    transpose_pad_kernel<<<dim3((unsigned)((n3 + 255) / 256)), 256, 0, stream>>>(W_out, WoutT, E_, DIM, EPAD, NP_OUT);
  }
  // 3) GEMM1: xp = xn @ W_in + b_in   (M=32768, N=336, K=512)
  gemm_bf16_wmma<<<dim3(NP_IN / 64, NTOK / 64), 128, 0, stream>>>(xn_h, WinT, b_in, xp, NTOK, E_, DIM);
  // 4) L2 row norm -> bf16 padded
  rownorm_kernel<<<dim3(NTOK), 32, 0, stream>>>(xp, xpn_h);
  // 5) GEMM2: qkv = xpn @ W_qkv + b_qkv   (M=32768, N=1008, K=384)
  gemm_bf16_wmma<<<dim3(NP_QKV / 64, NTOK / 64), 128, 0, stream>>>(xpn_h, WqkvT, b_qkv, qkv, NTOK, QKV3, EPAD);
  // 6) top-1 head attention
  {
    long n = (long)NTOK * HEADS;
    attn_kernel<<<dim3((unsigned)((n + 255) / 256)), 256, 0, stream>>>(qkv, attn_o);
  }
  // 7) importance + per-batch top-KC
  imp_kernel<<<dim3(NTOK), 32, 0, stream>>>(attn_o, imp);
  topk_kernel<<<dim3(B_), 256, 0, stream>>>(imp, idx);
  // 8) fused conv4 + SSM scan -> ys (bf16, padded)
  scan_kernel<<<dim3(B_), EPAD, 0, stream>>>(attn_o, idx, convw, A, Bp, Cp, ys_h);
  // 9) GEMM3: xproc = ys @ W_out + b_out   (M=B*KCPAD=6656, N=512, K=384)
  gemm_bf16_wmma<<<dim3(NP_OUT / 64, (B_ * KCPAD) / 64), 128, 0, stream>>>(ys_h, WoutT, b_out, xproc, B_ * KCPAD, DIM, EPAD);
  // 10) residual copy + scatter
  {
    long n = (long)NTOK * DIM;
    copy_kernel<<<dim3((unsigned)((n + 255) / 256)), 256, 0, stream>>>(x, out, n);
    long m = (long)B_ * KC * DIM;
    scatter_kernel<<<dim3((unsigned)((m + 255) / 256)), 256, 0, stream>>>(xproc, idx, x, out);
  }
}
